// Margin_Loss_38465727103368
// MI455X (gfx1250) — matrix operand order repaired and checked by reading, not compiled, not run on hardware
//
#include <hip/hip_runtime.h>
#include <hip/hip_bf16.h>

typedef __attribute__((ext_vector_type(16))) _Float16 v16h;
typedef __attribute__((ext_vector_type(8)))  _Float16 v8h;
typedef __attribute__((ext_vector_type(8)))  float    v8f;
typedef __attribute__((ext_vector_type(4)))  float    v4f;

#define NROWS 8192
#define DIM   128
#define KCLS  8
#define NPAIR (NROWS * (KCLS - 1))   // 57344

#define LN2F 0.69314718f

// ---------------------------------------------------------------------------
// Kernel 0: zero the scalar accumulators (sum_loss, pair_cnt, sum_beta)
// ---------------------------------------------------------------------------
__global__ void ml_zero_kernel(float* acc) {
    if (threadIdx.x < 4) acc[threadIdx.x] = 0.0f;
}

// ---------------------------------------------------------------------------
// Kernel 1: convert x (f32) -> x16 (f16) and compute per-row squared norms.
// One block (128 threads) per row.
// ---------------------------------------------------------------------------
__global__ void ml_prep_kernel(const float* __restrict__ x,
                               _Float16* __restrict__ x16,
                               float* __restrict__ sq) {
    __shared__ float red[128];
    const int row = blockIdx.x;
    const int t   = threadIdx.x;
    float v = x[row * DIM + t];
    x16[row * DIM + t] = (_Float16)v;
    red[t] = v * v;
    __syncthreads();
    #pragma unroll
    for (int s = 64; s > 0; s >>= 1) {
        if (t < s) red[t] += red[t + s];
        __syncthreads();
    }
    if (t == 0) sq[row] = red[0];
}

// ---------------------------------------------------------------------------
// Kernel 2: fused Gram (WMMA f16->f32) + distance-weighted Gumbel-max sampling.
// Grid: 64 blocks x 256 threads (8 waves). Each wave owns a 16-row stripe and
// streams 512 16-column tiles. Per tile: 4 x v_wmma_f32_16x16x32_f16 (K=128),
// then the sampling epilogue updates 7 running argmax keys per C-slot.
// ---------------------------------------------------------------------------
__global__ __launch_bounds__(256, 1)
void ml_gram_sample_kernel(const _Float16* __restrict__ x16,
                           const float* __restrict__ sq,
                           int* __restrict__ n_index) {
    // 8 waves * 32 lanes * 8 slots * 7 samples * 4B = 57344 B
    __shared__ unsigned int bestsLds[8 * 32 * 8 * 7];

    const int lane    = threadIdx.x & 31;
    const int wave    = threadIdx.x >> 5;
    const int rowTile = blockIdx.x * 8 + wave;   // 0..511
    const int rowBase = rowTile * 16;
    const int laneHi  = lane >> 4;               // 0/1
    const int laneLo  = lane & 15;

    // ---- A tiles (16x32 f16, ISA per-lane layout), K = feature dim 0..127 ----
    v16h A[4];
    {
        const _Float16* arow = x16 + (size_t)(rowBase + laneLo) * DIM;
        #pragma unroll
        for (int kb = 0; kb < 4; ++kb) {
            union { v16h v; v8h h[2]; } u;
            u.h[0] = *(const v8h*)(arow + kb * 32 + laneHi * 8);
            u.h[1] = *(const v8h*)(arow + kb * 32 + 16 + laneHi * 8);
            A[kb] = u.v;
        }
    }

    // squared norms for the 8 C-matrix row slots of this lane
    float sqr[8];
    #pragma unroll
    for (int r = 0; r < 8; ++r) sqr[r] = sq[rowBase + r + laneHi * 8];

    // running Gumbel-max keys: (monotone-float top 19 bits | col 13 bits)
    unsigned int best[7][8];
    #pragma unroll
    for (int s = 0; s < 7; ++s)
        #pragma unroll
        for (int r = 0; r < 8; ++r) best[s][r] = 0u;

    // per-sample salts (compile-time immediates inside the unrolled loop)
    const unsigned int SALT[7] = {
        0x243F6A88u, 0x85A308D3u, 0x13198A2Eu, 0x03707344u,
        0xA4093822u, 0x299F31D0u, 0x082EFA98u
    };

    for (int jt = 0; jt < 512; ++jt) {
        const int colBase = jt * 16;
        const _Float16* brow = x16 + (size_t)(colBase + laneLo) * DIM + laneHi * 8;
        // prefetch next tile's B stripe (global_prefetch_b8)
        __builtin_prefetch(brow + DIM * 16, 0, 3);

        v8f c = {};
        #pragma unroll
        for (int kb = 0; kb < 4; ++kb) {
            union { v16h v; v8h h[2]; } u;
            u.h[0] = *(const v8h*)(brow + kb * 32);
            u.h[1] = *(const v8h*)(brow + kb * 32 + 16);
            c = __builtin_amdgcn_wmma_f32_16x16x32_f16(
                    false, A[kb], false, u.v, (short)0, c, false, false);
        }

        const int   col = colBase + laneLo;
        const float sqc = sq[col];

        #pragma unroll
        for (int r = 0; r < 8; ++r) {
            const int row = rowBase + r + laneHi * 8;
            const float d2 = sqr[r] + sqc - 2.0f * c[r];
            // mask: different K-block and dist < 1.4  (dist^2 < 1.96; the
            // max(dist,0.5) clamp never flips this test)
            if (((row >> 3) != (col >> 3)) && (d2 < 1.96f)) {
                float dist = __builtin_amdgcn_sqrtf(fmaxf(d2, 0.0f));
                dist = fmaxf(dist, 0.5f);
                // (2-d)*ln(dist) - (d-3)/2*ln(1-dist^2/4), d=128, via log2:
                // -126*ln2^-1 folded: -126*ln(x) = -87.336544*log2(x), etc.
                const float logw =
                      -87.336544f * __builtin_amdgcn_logf(dist)
                    - 43.321699f * __builtin_amdgcn_logf(1.0f - 0.25f * dist * dist);
                // one full-avalanche mix per element ...
                unsigned int h0 = (unsigned)row * 8192u + (unsigned)col;
                h0 ^= h0 >> 16; h0 *= 0x7feb352du;
                h0 ^= h0 >> 15; h0 *= 0x846ca68bu;
                h0 ^= h0 >> 16;
                #pragma unroll
                for (int s = 0; s < 7; ++s) {
                    // ... then a cheap 3-op per-sample stream derivation
                    unsigned int h = (h0 ^ SALT[s]) * 0x9E3779B1u;
                    h ^= h >> 16;
                    const float uu = (float)(h >> 8) * 5.9604645e-08f + 1e-9f;
                    const float e  = -LN2F * __builtin_amdgcn_logf(uu);  // Exp(1)
                    const float g  = -LN2F * __builtin_amdgcn_logf(e);   // Gumbel
                    const float v  = logw + g;
                    unsigned int uv = __float_as_uint(v);
                    unsigned int mo = ((int)uv < 0) ? ~uv : (uv | 0x80000000u);
                    const unsigned int key = (mo & 0xFFFFE000u) | (unsigned)col;
                    if (key > best[s][r]) best[s][r] = key;
                }
            }
        }
    }

    // ---- cross-lane argmax: row m<8 lives in lanes 0-15 slot m; m>=8 in
    // lanes 16-31 slot m-8. Reduce via LDS. ----
    #pragma unroll
    for (int s = 0; s < 7; ++s)
        #pragma unroll
        for (int r = 0; r < 8; ++r)
            bestsLds[(((wave * 32 + lane) * 8) + r) * 7 + s] = best[s][r];
    __syncthreads();

    for (int t = threadIdx.x; t < 8 * 16 * 7; t += 256) {
        const int wv   = t / (16 * 7);
        const int rem  = t % (16 * 7);
        const int mrow = rem / 7;
        const int s    = rem % 7;
        const int laneBase = wv * 32 + ((mrow >= 8) ? 16 : 0);
        const int slot = mrow & 7;
        unsigned int mx = 0u;
        #pragma unroll 4
        for (int l = 0; l < 16; ++l) {
            const unsigned int v = bestsLds[((laneBase + l) * 8 + slot) * 7 + s];
            if (v > mx) mx = v;
        }
        const int grow = (blockIdx.x * 8 + wv) * 16 + mrow;
        n_index[grow * 7 + s] = (int)(mx & 0x1FFFu);
    }
}

// ---------------------------------------------------------------------------
// Kernel 3: triplet margin loss over 57344 (anchor, positive, negative) pairs.
// One thread per pair, f32 float4 loads, LDS block reduce + f32 atomics.
// ---------------------------------------------------------------------------
__global__ void ml_pair_loss_kernel(const float* __restrict__ x,
                                    const float* __restrict__ beta,
                                    const int* __restrict__ n_index,
                                    float* __restrict__ acc) {
    const int i = blockIdx.x * blockDim.x + threadIdx.x;
    float loss = 0.0f, cnt = 0.0f, bsum = 0.0f;
    if (i < NPAIR) {
        const int a   = i / 7;
        const int j   = i % 7;
        const int blk = a >> 3;
        const int ai  = a & 7;
        const int p   = (blk << 3) + ((j < ai) ? j : j + 1);
        const int nn  = n_index[i];
        const v4f* xa = (const v4f*)(x + (size_t)a  * DIM);
        const v4f* xp = (const v4f*)(x + (size_t)p  * DIM);
        const v4f* xn = (const v4f*)(x + (size_t)nn * DIM);
        float s_ap = 0.0f, s_an = 0.0f;
        #pragma unroll 8
        for (int q = 0; q < DIM / 4; ++q) {
            const v4f va = xa[q], vp = xp[q], vn = xn[q];
            const v4f dp = va - vp, dn = va - vn;
            s_ap += dp[0]*dp[0] + dp[1]*dp[1] + dp[2]*dp[2] + dp[3]*dp[3];
            s_an += dn[0]*dn[0] + dn[1]*dn[1] + dn[2]*dn[2] + dn[3]*dn[3];
        }
        const float bw   = beta[a];
        const float d_ap = __builtin_amdgcn_sqrtf(s_ap + 1e-8f);
        const float d_an = __builtin_amdgcn_sqrtf(s_an + 1e-8f);
        const float pl = fmaxf(d_ap - bw + 0.2f, 0.0f);
        const float nl = fmaxf(bw - d_an + 0.2f, 0.0f);
        loss = pl + nl;
        cnt  = ((pl > 0.0f) ? 1.0f : 0.0f) + ((nl > 0.0f) ? 1.0f : 0.0f);
        bsum = bw;
    }
    __shared__ float rl[256], rc[256], rb[256];
    rl[threadIdx.x] = loss; rc[threadIdx.x] = cnt; rb[threadIdx.x] = bsum;
    __syncthreads();
    #pragma unroll
    for (int s = 128; s > 0; s >>= 1) {
        if (threadIdx.x < s) {
            rl[threadIdx.x] += rl[threadIdx.x + s];
            rc[threadIdx.x] += rc[threadIdx.x + s];
            rb[threadIdx.x] += rb[threadIdx.x + s];
        }
        __syncthreads();
    }
    if (threadIdx.x == 0) {
        atomicAdd(&acc[0], rl[0]);
        atomicAdd(&acc[1], rc[0]);
        atomicAdd(&acc[2], rb[0]);
    }
}

// ---------------------------------------------------------------------------
// Kernel 4: finalize  loss = (sum + NU*sum_beta) / pair_cnt   (NU = 0)
// ---------------------------------------------------------------------------
__global__ void ml_finalize_kernel(const float* __restrict__ acc,
                                   float* __restrict__ out) {
    if (threadIdx.x == 0 && blockIdx.x == 0) {
        out[0] = (acc[0] + 0.0f * acc[2]) / acc[1];
    }
}

// ---------------------------------------------------------------------------
extern "C" void kernel_launch(void* const* d_in, const int* in_sizes, int n_in,
                              void* d_out, int out_size, void* d_ws, size_t ws_size,
                              hipStream_t stream) {
    (void)in_sizes; (void)n_in; (void)out_size; (void)ws_size;
    const float* x    = (const float*)d_in[0];   // [8192,128] f32
    // d_in[1] = y (int64) — unused by the reference loss (blocks come from index//K)
    const float* beta = (const float*)d_in[2];   // [8192] f32

    char* ws = (char*)d_ws;
    _Float16* x16  = (_Float16*)ws;                                   // 2 MB
    float*    sq   = (float*)(ws + 2u * 1024u * 1024u);               // 32 KB
    int*      nidx = (int*)(ws + 2u * 1024u * 1024u + 32u * 1024u);   // 224 KB
    float*    acc  = (float*)(ws + 2u * 1024u * 1024u + 288u * 1024u);

    ml_zero_kernel<<<1, 32, 0, stream>>>(acc);
    ml_prep_kernel<<<NROWS, 128, 0, stream>>>(x, x16, sq);
    ml_gram_sample_kernel<<<64, 256, 0, stream>>>(x16, sq, nidx);
    ml_pair_loss_kernel<<<(NPAIR + 255) / 256, 256, 0, stream>>>(x, beta, nidx, acc);
    ml_finalize_kernel<<<1, 1, 0, stream>>>(acc, (float*)d_out);
}
